// SparseLinear_66005057405275
// MI455X (gfx1250) — matrix operand order repaired and checked by reading, not compiled, not run on hardware
//
#include <hip/hip_runtime.h>
#include <stdint.h>

typedef __attribute__((ext_vector_type(8)))  __bf16 v8bf;
typedef __attribute__((ext_vector_type(16))) __bf16 v16bf;
typedef __attribute__((ext_vector_type(8)))  float  v8f;

#define KDIM 4096
#define NDIM 4096
#define MBLK 128
#define NBLK 128
#define KBLK 64
#define STR  72          // padded LDS row stride in bf16 elems: 144B, 16B aligned
#define KSTEPS (KDIM / KBLK)

// split an f32 into bf16 hi (RNE) + bf16 lo (residual)
__device__ __forceinline__ void bf16_split(float v, __bf16& h, __bf16& l) {
    h = (__bf16)v;
    l = (__bf16)(v - (float)h);
}

__device__ __forceinline__ v16bf cat8(v8bf a, v8bf b) {
    return __builtin_shufflevector(a, b, 0,1,2,3,4,5,6,7,8,9,10,11,12,13,14,15);
}

__global__ __launch_bounds__(256, 2)
void sparse_linear_wmma(const float* __restrict__ x,
                        const float* __restrict__ w,
                        const unsigned char* __restrict__ mask,
                        float* __restrict__ out) {
    // ping-pong double buffers: 4 planes x 2 bufs x 128 x 72 bf16 = 144 KB
    __shared__ alignas(16) __bf16 ldsXh[2][MBLK * STR];
    __shared__ alignas(16) __bf16 ldsXl[2][MBLK * STR];
    __shared__ alignas(16) __bf16 ldsWh[2][NBLK * STR];
    __shared__ alignas(16) __bf16 ldsWl[2][NBLK * STR];

    const int tid   = threadIdx.x;
    const int lane  = tid & 31;
    const int wave  = tid >> 5;
    const int waveM = wave & 1;    // 0..1 -> which 64-row slab of M
    const int waveN = wave >> 1;   // 0..3 -> which 32-col slab of N

    const int m0 = blockIdx.y * MBLK;
    const int n0 = blockIdx.x * NBLK;

    const int lr = tid >> 2;        // 0..63: load row within a pass
    const int lc = (tid & 3) * 16;  // 16-element k chunk within KBLK

    v8f zero = {};
    v8f acc[4][2];
    #pragma unroll
    for (int i = 0; i < 4; ++i)
        #pragma unroll
        for (int j = 0; j < 2; ++j) acc[i][j] = zero;

    // stage one K-tile (f32 -> bf16 hi/lo planes) into LDS buffer `buf`
    auto stage = [&](int kt, int buf) {
        if (kt + KBLK < KDIM) {
            __builtin_prefetch(x + (size_t)(m0 + lr) * KDIM + kt + KBLK + lc, 0, 0);
            __builtin_prefetch(w + (size_t)(n0 + lr) * KDIM + kt + KBLK + lc, 0, 0);
        }
        // ---- x tile: 128 x 64 f32 ----
        #pragma unroll
        for (int pass = 0; pass < 2; ++pass) {
            const int row = pass * 64 + lr;
            const float4* src = (const float4*)(x + (size_t)(m0 + row) * KDIM + kt + lc);
            #pragma unroll
            for (int half = 0; half < 2; ++half) {
                v8bf hv, lv;
                #pragma unroll
                for (int q = 0; q < 2; ++q) {
                    float4 v = src[half * 2 + q];
                    float f[4] = { v.x, v.y, v.z, v.w };
                    #pragma unroll
                    for (int j = 0; j < 4; ++j) {
                        __bf16 h, l;
                        bf16_split(f[j], h, l);
                        hv[q * 4 + j] = h;
                        lv[q * 4 + j] = l;
                    }
                }
                *(v8bf*)&ldsXh[buf][row * STR + lc + half * 8] = hv;
                *(v8bf*)&ldsXl[buf][row * STR + lc + half * 8] = lv;
            }
        }
        // ---- w tile with mask: 128 x 64 f32 + 128 x 64 u8 ----
        #pragma unroll
        for (int pass = 0; pass < 2; ++pass) {
            const int row = pass * 64 + lr;
            const float4*   src = (const float4*)(w + (size_t)(n0 + row) * KDIM + kt + lc);
            const uint32_t* msk = (const uint32_t*)(mask + (size_t)(n0 + row) * KDIM + kt + lc);
            #pragma unroll
            for (int half = 0; half < 2; ++half) {
                v8bf hv, lv;
                #pragma unroll
                for (int q = 0; q < 2; ++q) {
                    float4   v  = src[half * 2 + q];
                    uint32_t mm = msk[half * 2 + q];
                    float f[4] = { v.x, v.y, v.z, v.w };
                    #pragma unroll
                    for (int j = 0; j < 4; ++j) {
                        float fv = ((mm >> (8 * j)) & 0xffu) ? f[j] : 0.0f;
                        __bf16 h, l;
                        bf16_split(fv, h, l);
                        hv[q * 4 + j] = h;
                        lv[q * 4 + j] = l;
                    }
                }
                *(v8bf*)&ldsWh[buf][row * STR + lc + half * 8] = hv;
                *(v8bf*)&ldsWl[buf][row * STR + lc + half * 8] = lv;
            }
        }
    };

    stage(0, 0);

    for (int i = 0; i < KSTEPS; ++i) {
        const int buf = i & 1;
        // guarantees: stage(i) complete AND compute(i-1) (same parity as buf^1) drained
        __syncthreads();
        if (i + 1 < KSTEPS) stage((i + 1) * KBLK, buf ^ 1);

        // ---- compute tile i: 2 k-substeps of 32, 4x2 tiles, 3 split terms ----
        #pragma unroll
        for (int ks = 0; ks < 2; ++ks) {
            // A fragments (16x32 bf16): two b128 per lane, per ISA layout
            const int arow = waveM * 64 + (lane & 15);
            const int acol = ks * 32 + ((lane >> 4) * 8);
            v16bf Ah[4], Al[4];
            #pragma unroll
            for (int tm = 0; tm < 4; ++tm) {
                const int base = (arow + tm * 16) * STR + acol;
                Ah[tm] = cat8(*(const v8bf*)&ldsXh[buf][base],
                              *(const v8bf*)&ldsXh[buf][base + 16]);
                Al[tm] = cat8(*(const v8bf*)&ldsXl[buf][base],
                              *(const v8bf*)&ldsXl[buf][base + 16]);
            }
            // B fragments (32x16 bf16): two b128 per lane, per ISA layout
            const int brow = waveN * 32 + (lane & 15);
            const int bcol = ks * 32 + ((lane >> 4) * 8);
            v16bf Bh[2], Bl[2];
            #pragma unroll
            for (int tn = 0; tn < 2; ++tn) {
                const int base = (brow + tn * 16) * STR + bcol;
                Bh[tn] = cat8(*(const v8bf*)&ldsWh[buf][base],
                              *(const v8bf*)&ldsWh[buf][base + 16]);
                Bl[tn] = cat8(*(const v8bf*)&ldsWl[buf][base],
                              *(const v8bf*)&ldsWl[buf][base + 16]);
            }
            #pragma unroll
            for (int tm = 0; tm < 4; ++tm)
                #pragma unroll
                for (int tn = 0; tn < 2; ++tn) {
                    acc[tm][tn] = __builtin_amdgcn_wmma_f32_16x16x32_bf16(
                        false, Ah[tm], false, Bh[tn], (short)0, acc[tm][tn], false, false);
                    acc[tm][tn] = __builtin_amdgcn_wmma_f32_16x16x32_bf16(
                        false, Ah[tm], false, Bl[tn], (short)0, acc[tm][tn], false, false);
                    acc[tm][tn] = __builtin_amdgcn_wmma_f32_16x16x32_bf16(
                        false, Al[tm], false, Bh[tn], (short)0, acc[tm][tn], false, false);
                }
        }
    }

    // ---- epilogue: C/D layout -> VGPR r holds rows r (lanes 0-15) / r+8 (16-31)
    #pragma unroll
    for (int tm = 0; tm < 4; ++tm) {
        const int row0 = m0 + waveM * 64 + tm * 16 + ((lane >> 4) * 8);
        #pragma unroll
        for (int tn = 0; tn < 2; ++tn) {
            const int col = n0 + waveN * 32 + tn * 16 + (lane & 15);
            #pragma unroll
            for (int r = 0; r < 8; ++r)
                out[(size_t)(row0 + r) * NDIM + col] = acc[tm][tn][r];
        }
    }
}

extern "C" void kernel_launch(void* const* d_in, const int* in_sizes, int n_in,
                              void* d_out, int out_size, void* d_ws, size_t ws_size,
                              hipStream_t stream) {
    (void)n_in; (void)out_size; (void)d_ws; (void)ws_size;
    const float*         x    = (const float*)d_in[0];
    const float*         wt   = (const float*)d_in[1];
    const unsigned char* mask = (const unsigned char*)d_in[2];
    float*               out  = (float*)d_out;

    const int M = in_sizes[0] / KDIM;   // B*S = 8192
    dim3 grid(NDIM / NBLK, M / MBLK);   // (32, 64)
    dim3 block(256);
    sparse_linear_wmma<<<grid, block, 0, stream>>>(x, wt, mask, out);
}
